// TimeMix_28853590294664
// MI455X (gfx1250) — matrix operand order repaired and checked by reading, not compiled, not run on hardware
//
#include <hip/hip_runtime.h>
#include <hip/hip_bf16.h>
#include <math.h>

// Problem dims (fixed by the reference)
#define B_ 128
#define T_ 512
#define F_ 512
#define M_ 512

typedef __bf16 bf16_t;
typedef __attribute__((ext_vector_type(8)))  __bf16 v8bf;
typedef __attribute__((ext_vector_type(16))) __bf16 v16bf;
typedef __attribute__((ext_vector_type(8)))  float  v8f;

// ---------------------------------------------------------------------------
// Kernel 1: LayerNorm statistics over the T axis for each (b,f) column.
// Threads map to consecutive f -> every global load is fully coalesced.
// ---------------------------------------------------------------------------
__global__ __launch_bounds__(256)
void ln_stats_kernel(const float* __restrict__ x,
                     float* __restrict__ mean, float* __restrict__ rstd) {
    int idx = blockIdx.x * blockDim.x + threadIdx.x;   // over B*F
    if (idx >= B_ * F_) return;
    int b = idx / F_;
    int f = idx - b * F_;
    const float* p = x + (size_t)b * T_ * F_ + f;
    float s = 0.f, s2 = 0.f;
#pragma unroll 8
    for (int t = 0; t < T_; ++t) {
        float v = p[(size_t)t * F_];
        s  += v;
        s2 += v * v;
    }
    float m   = s * (1.0f / T_);
    float var = fmaxf(s2 * (1.0f / T_) - m * m, 0.0f);
    mean[idx] = m;
    rstd[idx] = rsqrtf(var + 1e-5f);
}

// ---------------------------------------------------------------------------
// Kernel 2: normalize + affine (gamma/beta indexed by t) and transpose to
// bf16 Xn[B, F, T] via LDS so the GEMM's B-fragments are contiguous in K(=t).
// Block: 32x8 threads handling a 32(t) x 32(f) tile.
// ---------------------------------------------------------------------------
__global__ __launch_bounds__(256)
void ln_transpose_kernel(const float* __restrict__ x,
                         const float* __restrict__ gamma,
                         const float* __restrict__ beta,
                         const float* __restrict__ mean,
                         const float* __restrict__ rstd,
                         bf16_t* __restrict__ xn) {
    __shared__ float tile[32][33];          // +1 pad: no LDS bank conflicts
    int b  = blockIdx.z;
    int t0 = blockIdx.x * 32;
    int f0 = blockIdx.y * 32;
    int tx = threadIdx.x;                   // 0..31
    int ty = threadIdx.y;                   // 0..7
#pragma unroll
    for (int k = 0; k < 4; ++k) {
        int t = t0 + ty + 8 * k;
        int f = f0 + tx;
        float v  = x[(size_t)b * T_ * F_ + (size_t)t * F_ + f];
        float nm = (v - mean[b * F_ + f]) * rstd[b * F_ + f];
        tile[ty + 8 * k][tx] = nm * gamma[t] + beta[t];
    }
    __syncthreads();
#pragma unroll
    for (int k = 0; k < 4; ++k) {
        int f = f0 + ty + 8 * k;
        int t = t0 + tx;
        xn[(size_t)b * F_ * T_ + (size_t)f * T_ + t] = (bf16_t)tile[tx][ty + 8 * k];
    }
}

// ---------------------------------------------------------------------------
// Kernel 3: convert W [M,T] f32 -> bf16 (row-major already A-fragment friendly)
// ---------------------------------------------------------------------------
__global__ __launch_bounds__(256)
void w_convert_kernel(const float* __restrict__ W, bf16_t* __restrict__ Wb) {
    int idx = blockIdx.x * blockDim.x + threadIdx.x;
    if (idx < M_ * T_) Wb[idx] = (bf16_t)W[idx];
}

// ---------------------------------------------------------------------------
// Kernel 4: per-batch GEMM  Y[b] = W (M x T) * Xn[b]^T-layout (T x F),
// bias + exact GELU + residual, all through v_wmma_f32_16x16x32_bf16.
//
// Block = 128 threads = 4 waves; each wave computes a 64x64 output block
// (4x4 WMMA tiles). Block covers 128x128 of the 512x512 output.
//
// Fragment layouts (wave32; A/B in this code's contiguous K order, C/D per CDNA5 ISA 7.12.2):
//  A (16x32 bf16): lane l (l&15 = row m), low 4 VGPRs = K kb..kb+7,
//                  high 4 VGPRs = K kb+8..kb+15, kb = k0 + 16*(l>>4).
//  B (32x16 bf16): lane l (l&15 = col f), 8 VGPRs = 16 consecutive K
//                  starting at k0 + 16*(l>>4).
//  C/D (16x16 f32): VGPR r, lane l -> m = r + 8*(l>>4), f = l&15.
// ---------------------------------------------------------------------------
__global__ __launch_bounds__(128)
void gemm_gelu_kernel(const bf16_t* __restrict__ Wb,
                      const bf16_t* __restrict__ Xn,
                      const float* __restrict__ bias,
                      const float* __restrict__ x,
                      float* __restrict__ out) {
    int b    = blockIdx.z;
    int lane = threadIdx.x & 31;
    int wave = threadIdx.x >> 5;                 // 0..3
    int mBase = blockIdx.y * 128 + (wave >> 1) * 64;
    int fBase = blockIdx.x * 128 + (wave & 1)  * 64;

    int lgrp = lane >> 4;                        // 0 or 1 (lane half)
    int lmod = lane & 15;

    const bf16_t* Xb = Xn + (size_t)b * F_ * T_;

    v8f acc[4][4];
#pragma unroll
    for (int i = 0; i < 4; ++i)
#pragma unroll
        for (int j = 0; j < 4; ++j)
#pragma unroll
            for (int r = 0; r < 8; ++r)
                acc[i][j][r] = 0.0f;

    for (int k0 = 0; k0 < T_; k0 += 32) {
        // gfx1250 prefetch of next K-slab (emits global_prefetch_b8)
        if (k0 + 32 < T_) {
            __builtin_prefetch(Xb + (size_t)(fBase + lmod) * T_ + (k0 + 32), 0, 3);
            __builtin_prefetch(Wb + (size_t)(mBase + lmod) * T_ + (k0 + 32), 0, 3);
        }

        v16bf afrag[4];
        v16bf bfrag[4];
#pragma unroll
        for (int i = 0; i < 4; ++i) {
            int m  = mBase + 16 * i + lmod;
            int kb = k0 + 16 * lgrp;
            const bf16_t* p = Wb + (size_t)m * T_ + kb;
            v8bf lo = *(const v8bf*)(p);          // K kb..kb+7
            v8bf hi = *(const v8bf*)(p + 8);      // K kb+8..kb+15
            afrag[i] = __builtin_shufflevector(lo, hi,
                0, 1, 2, 3, 4, 5, 6, 7, 8, 9, 10, 11, 12, 13, 14, 15);
        }
#pragma unroll
        for (int j = 0; j < 4; ++j) {
            int f  = fBase + 16 * j + lmod;
            int kb = k0 + 16 * lgrp;
            bfrag[j] = *(const v16bf*)(Xb + (size_t)f * T_ + kb);  // 32B contiguous
        }
#pragma unroll
        for (int i = 0; i < 4; ++i)
#pragma unroll
            for (int j = 0; j < 4; ++j)
                acc[i][j] = __builtin_amdgcn_wmma_f32_16x16x32_bf16(
                    false, afrag[i], false, bfrag[j],
                    (short)0, acc[i][j], false, false);
    }

    // Epilogue: bias + exact GELU + residual (out[b,m,f] += x[b,m,f], M == T)
#pragma unroll
    for (int i = 0; i < 4; ++i) {
#pragma unroll
        for (int j = 0; j < 4; ++j) {
            int f = fBase + 16 * j + lmod;
#pragma unroll
            for (int r = 0; r < 8; ++r) {
                int m = mBase + 16 * i + r + 8 * lgrp;
                float z = acc[i][j][r] + bias[m];
                float g = 0.5f * z * (1.0f + erff(z * 0.70710678118654752f));
                size_t o = (size_t)b * M_ * F_ + (size_t)m * F_ + f;
                out[o] = g + x[o];
            }
        }
    }
}

// ---------------------------------------------------------------------------
// Host-side launch (graph-capture safe: only kernel launches on `stream`).
// Workspace layout: Wb (512KB) | mean (256KB) | rstd (256KB) | Xn (64MB)
// ---------------------------------------------------------------------------
extern "C" void kernel_launch(void* const* d_in, const int* in_sizes, int n_in,
                              void* d_out, int out_size, void* d_ws, size_t ws_size,
                              hipStream_t stream) {
    const float* x     = (const float*)d_in[0];
    const float* gamma = (const float*)d_in[1];
    const float* beta  = (const float*)d_in[2];
    const float* W     = (const float*)d_in[3];
    const float* bias  = (const float*)d_in[4];
    float* out = (float*)d_out;

    char* ws = (char*)d_ws;
    bf16_t* Wb   = (bf16_t*)(ws);
    float*  mean = (float*)(ws + (512u << 10));
    float*  rstd = (float*)(ws + (768u << 10));
    bf16_t* Xn   = (bf16_t*)(ws + (1024u << 10));

    ln_stats_kernel<<<(B_ * F_ + 255) / 256, 256, 0, stream>>>(x, mean, rstd);
    w_convert_kernel<<<(M_ * T_ + 255) / 256, 256, 0, stream>>>(W, Wb);

    dim3 tgrid(T_ / 32, F_ / 32, B_);
    ln_transpose_kernel<<<tgrid, dim3(32, 8), 0, stream>>>(x, gamma, beta, mean, rstd, Xn);

    dim3 ggrid(F_ / 128, M_ / 128, B_);
    gemm_gelu_kernel<<<ggrid, 128, 0, stream>>>(Wb, Xn, bias, x, out);
}
